// CohereAttention_64699387347274
// MI455X (gfx1250) — matrix operand order repaired and checked by reading, not compiled, not run on hardware
//
#include <hip/hip_runtime.h>
#include <hip/hip_bf16.h>

// ---------------------------------------------------------------------------
// CDNA5 (gfx1250) attention block: bf16 WMMA everywhere, fp32 accumulate.
// GEMMs stage tiles into LDS with the Tensor Data Mover (TDM) when available.
// ---------------------------------------------------------------------------

typedef __bf16 bf16_t;
typedef __attribute__((ext_vector_type(16))) __bf16 v16bf;
typedef __attribute__((ext_vector_type(8)))  float  v8f;
typedef __attribute__((ext_vector_type(4)))  unsigned int v4u;
typedef __attribute__((ext_vector_type(8)))  int v8i_t;
typedef __attribute__((ext_vector_type(4)))  int v4i_t;

#if defined(__has_builtin)
#if __has_builtin(__builtin_amdgcn_tensor_load_to_lds) && \
    __has_builtin(__builtin_amdgcn_s_wait_tensorcnt)
#define USE_TDM 1
#endif
#endif
#ifndef USE_TDM
#define USE_TDM 0
#endif

union Frag16 {
    v16bf        v;
    unsigned int u32[8];
    uint4        u128[2];
};

// A-matrix 16x32 bf16 fragment (ISA 7.12.2): lane 0-15 row=lane, K pairs
// {0..7,16..23}; lane 16-31 row=lane-16, K pairs {8..15,24..31}.
__device__ __forceinline__ v16bf load_a_frag(const bf16_t* p, int ld, int row0,
                                             int k0, int lane) {
    Frag16 f;
    const bf16_t* rp =
        p + (size_t)(row0 + (lane & 15)) * ld + k0 + ((lane >> 4) << 3);
#pragma unroll
    for (int i = 0; i < 4; ++i) {
        f.u32[i]     = *reinterpret_cast<const unsigned int*>(rp + 2 * i);
        f.u32[i + 4] = *reinterpret_cast<const unsigned int*>(rp + 16 + 2 * i);
    }
    return f.v;
}

// B-matrix 32x16 bf16 fragment: lane holds column n = lane&15, K values
// {0..7,16..23} (lanes 0-15) / {8..15,24..31} (lanes 16-31). Element (k,n) = src[(n0+n)*ld + k0 + k].
__device__ __forceinline__ v16bf load_b_frag(const bf16_t* p, int ld, int n0,
                                             int k0, int lane) {
    Frag16 f;
    const bf16_t* sp =
        p + (size_t)(n0 + (lane & 15)) * ld + k0 + ((lane >> 4) << 3);
    f.u128[0] = *reinterpret_cast<const uint4*>(sp);
    f.u128[1] = *reinterpret_cast<const uint4*>(sp + 16);
    return f.v;
}

__device__ __forceinline__ v8f wmma_bf16(v16bf a, v16bf b, v8f c) {
    return __builtin_amdgcn_wmma_f32_16x16x32_bf16(
        false, a, false, b, (short)0, c, false, false);
}

#if USE_TDM
// ---------------------------------------------------------------------------
// TDM: DMA a (64 rows x 32 cols) bf16 tile from a row-major matrix into LDS.
// D# built per CDNA5 ISA:
//   group0: count=1 | lds_addr[63:32] | global_addr[120:64] | type=2
//   group1: data_size=1(2B) | tensor_dim0/1 | tile_dim0=32, tile_dim1=64 |
//           tensor_dim0_stride=ld | tensor_dim1_stride=ld
// groups 2/3 + trailing group zero (2D tile). Tracked on TENSORcnt.
// This toolchain exposes the 6-arg builtin:
//   (uint32x4, int32x8, int32x4, int32x4, int32x8, i32 cpol)
// ---------------------------------------------------------------------------
__device__ __forceinline__ void tdm_load_tile(unsigned lds_off,
                                              const bf16_t* gsrc,
                                              unsigned tensor_rows,
                                              unsigned ld_elems) {
    union { unsigned long long q[2]; v4u v; } g0;
    union { unsigned long long q[4]; v8i_t v; } g1;
    unsigned long long ga  = (unsigned long long)(uintptr_t)gsrc;
    unsigned long long td0 = ld_elems;      // tensor dim0 (elements per row)
    unsigned long long td1 = tensor_rows;   // tensor dim1 (rows)
    unsigned long long st0 = ld_elems;      // dim0 stride (elements)

    g0.q[0] = 1ull | ((unsigned long long)lds_off << 32);
    g0.q[1] = (ga & 0x01FFFFFFFFFFFFFFull) | (2ull << 62);  // type=2 "image"

    g1.q[0] = (1ull << 16)                      // data_size = 1 -> 2 bytes
            | ((td0 & 0xFFFFull) << 48);        // tensor_dim0[15:0]
    g1.q[1] = ((td0 >> 16) & 0xFFFFull)         // tensor_dim0[31:16]
            | ((td1 & 0xFFFFFFFFull) << 16)     // tensor_dim1
            | (32ull << 48);                    // tile_dim0 = 32 elements
    g1.q[2] = 64ull                             // tile_dim1 = 64 rows
            | ((st0 & 0xFFFFFFFFull) << 32);    // dim0_stride[31:0]
    g1.q[3] = ((st0 >> 32) & 0xFFFFull)         // dim0_stride[47:32]
            | ((st0 & 0xFFFFFFFFFFFFull) << 16);// dim1_stride (unused here)

    v4i_t z4 = {0, 0, 0, 0};
    v8i_t z8 = {0, 0, 0, 0, 0, 0, 0, 0};
    __builtin_amdgcn_tensor_load_to_lds(g0.v, g1.v, z4, z4, z8, 0);
}
#else
// Fallback: cooperative LDS staging (256 threads copy 64x32 bf16 = 4KB).
__device__ __forceinline__ void coop_load_tile(bf16_t* dst, const bf16_t* src,
                                               int ld, int tid) {
    int row = tid >> 2;
    int col = (tid & 3) * 8;
    *reinterpret_cast<uint4*>(dst + row * 32 + col) =
        *reinterpret_cast<const uint4*>(src + (size_t)row * ld + col);
}
#endif

// ---------------------------------------------------------------------------
// fp32 -> bf16 streaming cast
// ---------------------------------------------------------------------------
__global__ void cast_f32_to_bf16(const float* __restrict__ x,
                                 bf16_t* __restrict__ y, long long n) {
    long long i = (long long)blockIdx.x * blockDim.x + threadIdx.x;
    long long stride = (long long)gridDim.x * blockDim.x;
    for (; i < n; i += stride) y[i] = (bf16_t)x[i];
}

// ---------------------------------------------------------------------------
// GEMM: Y(MxN, f32) = X(MxK, bf16, row-major) * W(NxK, bf16)^T
// Block 256 = 8 waves, tile 64x64. A/B 64x32 tiles double-buffered in LDS via
// TDM (wave 0 issues DMA, s_wait_tensorcnt overlaps DMA with WMMA).
// LDS layout (bytes): A0 @ 0, A1 @ 4096, B0 @ 8192, B1 @ 12288.
// NOTE: no pointer arrays built from the shared variable (lld can't encode
// addrspacecast in static initializers) -- scalar pointers + offset XOR only.
// ---------------------------------------------------------------------------
__global__ __launch_bounds__(256) void gemm_bf16_nt(
    const bf16_t* __restrict__ X, const bf16_t* __restrict__ W,
    float* __restrict__ Y, int M, int N, int K) {
    __shared__ __align__(128) char smem[4 * 4096];

    int tid  = threadIdx.x;
    int lane = tid & 31;
    int wave = tid >> 5;
    int mblk = blockIdx.y * 64;
    int nblk = blockIdx.x * 64;
    int aw = (wave & 3) * 16;   // wave's row offset in A tile
    int bw = (wave >> 2) * 32;  // wave's col offset in B tile

    v8f c0 = {};
    v8f c1 = {};
    int nk = K / 32;

#if USE_TDM
    // Flat LDS addresses carry the LDS byte offset in the low 32 bits
    // (ISA §10.2 aperture mapping) -> usable as D#.lds_addr.
    unsigned lds_base = (unsigned)(uintptr_t)(void*)smem;
    bool issuer = (wave == 0);
    if (issuer) {
        tdm_load_tile(lds_base, X + (size_t)mblk * K, M, K);
        tdm_load_tile(lds_base + 8192u, W + (size_t)nblk * K, N, K);
    }
    for (int ks = 0; ks < nk; ++ks) {
        unsigned aoff = (unsigned)((ks & 1) << 12);  // 0 or 4096 bytes
        if (issuer) {
            if (ks + 1 < nk) {
                int k2 = (ks + 1) * 32;
                tdm_load_tile(lds_base + (aoff ^ 4096u),
                              X + (size_t)mblk * K + k2, M, K);
                tdm_load_tile(lds_base + 8192u + (aoff ^ 4096u),
                              W + (size_t)nblk * K + k2, N, K);
                __builtin_amdgcn_s_wait_tensorcnt(2);  // current pair done
            } else {
                __builtin_amdgcn_s_wait_tensorcnt(0);
            }
        }
        __syncthreads();
        bf16_t* bufA = (bf16_t*)(smem + aoff);
        bf16_t* bufB = (bf16_t*)(smem + 8192u + aoff);
        v16bf a  = load_a_frag(bufA, 32, aw, 0, lane);
        v16bf b0 = load_b_frag(bufB, 32, bw, 0, lane);
        v16bf b1 = load_b_frag(bufB, 32, bw + 16, 0, lane);
        c0 = wmma_bf16(a, b0, c0);
        c1 = wmma_bf16(a, b1, c1);
        __syncthreads();
    }
#else
    for (int ks = 0; ks < nk; ++ks) {
        int k = ks * 32;
        bf16_t* bufA = (bf16_t*)(smem);
        bf16_t* bufB = (bf16_t*)(smem + 8192u);
        __syncthreads();
        coop_load_tile(bufA, X + (size_t)mblk * K + k, K, tid);
        coop_load_tile(bufB, W + (size_t)nblk * K + k, K, tid);
        __syncthreads();
        v16bf a  = load_a_frag(bufA, 32, aw, 0, lane);
        v16bf b0 = load_b_frag(bufB, 32, bw, 0, lane);
        v16bf b1 = load_b_frag(bufB, 32, bw + 16, 0, lane);
        c0 = wmma_bf16(a, b0, c0);
        c1 = wmma_bf16(a, b1, c1);
    }
#endif

    // D layout: lane 0-15 -> rows +0..7, lane 16-31 -> rows +8..15.
    int mrow = mblk + aw + ((lane >> 4) << 3);
    int ncol = nblk + bw + (lane & 15);
#pragma unroll
    for (int r = 0; r < 8; ++r) {
        Y[(size_t)(mrow + r) * N + ncol]      = c0[r];
        Y[(size_t)(mrow + r) * N + ncol + 16] = c1[r];
    }
}

// ---------------------------------------------------------------------------
// Fused per-head LayerNorm + RoPE, fp32 in -> bf16 out in (B, NH, S, D).
// ---------------------------------------------------------------------------
__global__ __launch_bounds__(128) void ln_rope_kernel(
    const float* __restrict__ X,      // (B*S, NH*128)
    const int* __restrict__ pos_ids,  // (S)
    const float* __restrict__ w,      // (NH, 128)
    bf16_t* __restrict__ Y,           // (B, NH, S, 128)
    int S, int NH) {
    int h   = blockIdx.x % NH;
    int tok = blockIdx.x / NH;  // b*S + s
    int s   = tok % S;
    int b   = tok / S;
    int d   = threadIdx.x;

    float x = X[(size_t)tok * (NH * 128) + h * 128 + d];
    __shared__ float red[128];

    red[d] = x;
    __syncthreads();
#pragma unroll
    for (int off = 64; off > 0; off >>= 1) {
        if (d < off) red[d] += red[d + off];
        __syncthreads();
    }
    float mean = red[0] * (1.0f / 128.0f);
    __syncthreads();

    float dx = x - mean;
    red[d] = dx * dx;
    __syncthreads();
#pragma unroll
    for (int off = 64; off > 0; off >>= 1) {
        if (d < off) red[d] += red[d + off];
        __syncthreads();
    }
    float var = red[0] * (1.0f / 128.0f);
    __syncthreads();

    float y = dx * rsqrtf(var + 1e-5f) * w[h * 128 + d];
    red[d] = y;
    __syncthreads();
    float partner = red[d ^ 64];  // rope pair: d<64 <-> d+64

    float pos = (float)pos_ids[s];
    int   i   = d & 63;
    float freq = __expf(-(float)i * (9.210340371976184f / 64.0f));
    float ang  = pos * freq;
    float c = __cosf(ang), sn = __sinf(ang);
    float out = (d < 64) ? (y * c - partner * sn) : (y * c + partner * sn);

    Y[(((size_t)(b * NH + h)) * S + s) * 128 + d] = (bf16_t)out;
}

// ---------------------------------------------------------------------------
// V: fp32 (B*S, KVH*128) -> bf16 (B, KVH, 128, S)  (D-major for PV B-frags)
// ---------------------------------------------------------------------------
__global__ void v_transpose_cast(const float* __restrict__ X,
                                 bf16_t* __restrict__ Y, int S) {
    size_t total = (size_t)2 * 8 * 128 * S;
    size_t idx = (size_t)blockIdx.x * blockDim.x + threadIdx.x;
    if (idx >= total) return;
    int    s = idx % S;
    size_t t = idx / S;
    int    d = t % 128;  t /= 128;
    int    h = t % 8;
    int    b = (int)(t / 8);
    Y[idx] = (bf16_t)X[((size_t)(b * S + s)) * (8 * 128) + h * 128 + d];
}

// ---------------------------------------------------------------------------
// Flash attention (causal, GQA 4:1). Block 128 = 4 waves; wave owns 16 q rows.
// ---------------------------------------------------------------------------
__global__ __launch_bounds__(128) void flash_attn_kernel(
    const bf16_t* __restrict__ Qb,  // (B, 32, S, 128)
    const bf16_t* __restrict__ Kb,  // (B, 8, S, 128)
    const bf16_t* __restrict__ Vt,  // (B, 8, 128, S)
    bf16_t* __restrict__ Ob,        // (B*S, 32*128)
    int S) {
    constexpr int H = 32, KVH = 8, D = 128;
    const float scale = 0.08838834764831845f;  // 1/sqrt(128)

    int lane = threadIdx.x & 31;
    int wave = threadIdx.x >> 5;
    int q0 = blockIdx.x * 64 + wave * 16;
    int h = blockIdx.y, b = blockIdx.z;
    int kvh = h >> 2;

    const bf16_t* Qp = Qb + ((size_t)(b * H + h) * S) * D;
    const bf16_t* Kp = Kb + ((size_t)(b * KVH + kvh) * S) * D;
    const bf16_t* Vp = Vt + ((size_t)(b * KVH + kvh) * D) * S;

    __shared__ bf16_t shP[4][16 * 32];
    bf16_t* Pw = &shP[wave][0];

    v16bf qa[4];
#pragma unroll
    for (int dc = 0; dc < 4; ++dc) qa[dc] = load_a_frag(Qp, D, q0, dc * 32, lane);

    v8f o[8] = {};
    float mstate[8], lstate[8];
#pragma unroll
    for (int r = 0; r < 8; ++r) { mstate[r] = -3.0e38f; lstate[r] = 0.0f; }

    int ncol  = lane & 15;
    int mbase = (lane >> 4) << 3;

    int nkb = (q0 + 47) >> 5;  // causal: key blocks of 32 covering keys<=q0+15
    for (int kb = 0; kb < nkb; ++kb) {
        int kbase = kb * 32;

        // ---- scores: S = Q * K^T (two 16x16 tiles over 32 keys)
        v8f s0 = {};
        v8f s1 = {};
#pragma unroll
        for (int dc = 0; dc < 4; ++dc) {
            v16bf k0f = load_b_frag(Kp, D, kbase,      dc * 32, lane);
            v16bf k1f = load_b_frag(Kp, D, kbase + 16, dc * 32, lane);
            s0 = wmma_bf16(qa[dc], k0f, s0);
            s1 = wmma_bf16(qa[dc], k1f, s1);
        }

        // ---- online softmax (row state per accumulator register)
        float p0[8], p1[8];
#pragma unroll
        for (int r = 0; r < 8; ++r) {
            int qrow = q0 + mbase + r;
            float v0 = s0[r] * scale;
            float v1 = s1[r] * scale;
            if (kbase + ncol > qrow) v0 = -3.0e38f;
            if (kbase + 16 + ncol > qrow) v1 = -3.0e38f;
            float rmax = fmaxf(v0, v1);
#pragma unroll
            for (int off = 8; off > 0; off >>= 1)
                rmax = fmaxf(rmax, __shfl_xor(rmax, off, 32));
            float mnew = fmaxf(mstate[r], rmax);
            float corr = __expf(mstate[r] - mnew);
            float e0 = __expf(v0 - mnew);
            float e1 = __expf(v1 - mnew);
            float rsum = e0 + e1;
#pragma unroll
            for (int off = 8; off > 0; off >>= 1)
                rsum += __shfl_xor(rsum, off, 32);
            lstate[r] = lstate[r] * corr + rsum;
            mstate[r] = mnew;
            p0[r] = e0;
            p1[r] = e1;
#pragma unroll
            for (int dt = 0; dt < 8; ++dt) o[dt][r] *= corr;
        }

        // ---- transpose P (D layout -> A layout) through per-wave LDS
#pragma unroll
        for (int r = 0; r < 8; ++r) {
            Pw[(mbase + r) * 32 + ncol]      = (bf16_t)p0[r];
            Pw[(mbase + r) * 32 + 16 + ncol] = (bf16_t)p1[r];
        }
        __builtin_amdgcn_wave_barrier();
        asm volatile("s_wait_dscnt 0" ::: "memory");
        v16bf pa = load_a_frag(Pw, 32, 0, 0, lane);
        __builtin_amdgcn_wave_barrier();

        // ---- PV: O += P * V  (8 d-tiles of 16)
#pragma unroll
        for (int dt = 0; dt < 8; ++dt) {
            v16bf vb = load_b_frag(Vp, S, dt * 16, kbase, lane);
            o[dt] = wmma_bf16(pa, vb, o[dt]);
        }
    }

    // ---- epilogue: normalize and store bf16 (B*S, H*D) for output GEMM
#pragma unroll
    for (int dt = 0; dt < 8; ++dt) {
#pragma unroll
        for (int r = 0; r < 8; ++r) {
            int qrow = q0 + mbase + r;
            float val = o[dt][r] / lstate[r];
            Ob[((size_t)(b * S + qrow)) * (H * D) + h * D + dt * 16 + ncol] =
                (bf16_t)val;
        }
    }
}

// ---------------------------------------------------------------------------
// Host-side orchestration
// ---------------------------------------------------------------------------
extern "C" void kernel_launch(void* const* d_in, const int* in_sizes, int n_in,
                              void* d_out, int out_size, void* d_ws,
                              size_t ws_size, hipStream_t stream) {
    (void)in_sizes; (void)n_in; (void)out_size; (void)ws_size;

    const float* hidden = (const float*)d_in[0];
    const int*   pos    = (const int*)d_in[1];
    const float* wq     = (const float*)d_in[2];
    const float* wk     = (const float*)d_in[3];
    const float* wv     = (const float*)d_in[4];
    const float* wo     = (const float*)d_in[5];
    const float* qnw    = (const float*)d_in[6];
    const float* knw    = (const float*)d_in[7];
    float*       out    = (float*)d_out;

    constexpr int B = 2, S = 2048, HID = 4096, NH = 32, NKV = 8;
    const size_t T = (size_t)B * S;  // 4096 tokens

    char*  ws  = (char*)d_ws;
    size_t off = 0;
    auto alloc = [&](size_t bytes) -> void* {
        off = (off + 255) & ~(size_t)255;
        void* p = ws + off;
        off += bytes;
        return p;
    };

    bf16_t* hb  = (bf16_t*)alloc(T * HID * 2);
    bf16_t* wqb = (bf16_t*)alloc((size_t)HID * HID * 2);
    bf16_t* wkb = (bf16_t*)alloc((size_t)1024 * HID * 2);
    bf16_t* wvb = (bf16_t*)alloc((size_t)1024 * HID * 2);
    bf16_t* wob = (bf16_t*)alloc((size_t)HID * HID * 2);
    float*  qf  = (float*)alloc(T * 4096 * 4);
    float*  kf  = (float*)alloc(T * 1024 * 4);
    float*  vf  = (float*)alloc(T * 1024 * 4);
    bf16_t* Qb  = (bf16_t*)alloc(T * 4096 * 2);
    bf16_t* Kb  = (bf16_t*)alloc(T * 1024 * 2);
    bf16_t* Vt  = (bf16_t*)alloc(T * 1024 * 2);
    bf16_t* Ob  = hb;  // hidden bf16 dead after the QKV GEMMs; alias.

    // 1) fp32 -> bf16 casts
    cast_f32_to_bf16<<<4096, 256, 0, stream>>>(hidden, hb, (long long)T * HID);
    cast_f32_to_bf16<<<4096, 256, 0, stream>>>(wq, wqb, (long long)HID * HID);
    cast_f32_to_bf16<<<2048, 256, 0, stream>>>(wk, wkb, (long long)1024 * HID);
    cast_f32_to_bf16<<<2048, 256, 0, stream>>>(wv, wvb, (long long)1024 * HID);
    cast_f32_to_bf16<<<4096, 256, 0, stream>>>(wo, wob, (long long)HID * HID);

    // 2) QKV projections (WMMA bf16, fp32 accumulate, TDM-staged tiles)
    gemm_bf16_nt<<<dim3(4096 / 64, T / 64), 256, 0, stream>>>(hb, wqb, qf,
                                                              (int)T, 4096, HID);
    gemm_bf16_nt<<<dim3(1024 / 64, T / 64), 256, 0, stream>>>(hb, wkb, kf,
                                                              (int)T, 1024, HID);
    gemm_bf16_nt<<<dim3(1024 / 64, T / 64), 256, 0, stream>>>(hb, wvb, vf,
                                                              (int)T, 1024, HID);

    // 3) per-head LayerNorm + RoPE -> bf16 head-major layouts
    ln_rope_kernel<<<(unsigned)(T * NH), 128, 0, stream>>>(qf, pos, qnw, Qb, S, NH);
    ln_rope_kernel<<<(unsigned)(T * NKV), 128, 0, stream>>>(kf, pos, knw, Kb, S, NKV);

    // 4) V -> (B, KVH, D, S) bf16
    {
        size_t total = (size_t)B * NKV * 128 * S;
        v_transpose_cast<<<(unsigned)((total + 255) / 256), 256, 0, stream>>>(vf, Vt, S);
    }

    // 5) causal flash attention (GQA 4:1), bf16 output in (B*S, 32*128)
    flash_attn_kernel<<<dim3(S / 64, NH, B), 128, 0, stream>>>(Qb, Kb, Vt, Ob, S);

    // 6) output projection -> fp32 d_out
    gemm_bf16_nt<<<dim3(4096 / 64, T / 64), 256, 0, stream>>>(Ob, wob, out,
                                                              (int)T, 4096, HID);
}
